// GaussianCNNGRUActor_11982958756360
// MI455X (gfx1250) — compile-verified, matrix operand order checked by reading, not run on hardware
//
#include <hip/hip_runtime.h>
#include <hip/hip_bf16.h>

// ---------------------------------------------------------------------------
// GaussianCNNGRUActor forward for MI455X (gfx1250, wave32, WMMA).
//   conv(3->6)->leaky->conv(6->1)->leaky  => seq [B,60,124]  (f16 in d_ws)
//   GRU(124->129, 60 steps) with v_wmma_f32_16x16x32_f16 tiles
//   FC heads (258->129) x2 with WMMA, then tanh-squashed Gaussian epilogue.
// Requires ws_size >= 4096*60*128*2 bytes (~60 MB) for the f16 seq buffer.
// ---------------------------------------------------------------------------

typedef _Float16 v16h __attribute__((ext_vector_type(16)));
typedef float    v8f  __attribute__((ext_vector_type(8)));

#define B_TOT 4096
#define T_SEQ 60
#define S_PAD 128      // 124 real + 4 zero pad (K pad to 128 = 4 k-tiles)
#define HID   129
#define G3    387      // 3*HID
#define NTILE 25       // ceil(387/16) gate column tiles
#define GPAD  400      // 25*16 padded gate width
#define HPAD  160      // 129 -> 5 k-tiles of 32
#define FPAD  288      // feat 258 -> 9 k-tiles of 32

// ---------------------------------------------------------------------------
// Kernel 1: fused conv1+leaky+conv2+leaky, one (batch, t-chunk) per block.
// ---------------------------------------------------------------------------
__global__ __launch_bounds__(256) void conv_kernel(
    const float* __restrict__ x,      // [B,3,64,128]
    const float* __restrict__ w1,     // [6,3,3,3]
    const float* __restrict__ b1,     // [6]
    const float* __restrict__ w2,     // [1,6,3,3]
    const float* __restrict__ b2,     // [1]
    _Float16* __restrict__ seq)       // [B,60,128] (cols 124..127 = 0)
{
  extern __shared__ char smem[];
  float* midL = (float*)smem;                 // 6*14*126 conv1 rows for chunk
  float* w1L  = midL + 6 * 14 * 126;          // 162
  float* b1L  = w1L + 162;                    // 6
  float* w2L  = b1L + 6;                      // 54
  float* b2L  = w2L + 54;                     // 1

  const int b   = blockIdx.x;
  const int t0  = blockIdx.y * 12;            // seq rows t0..t0+11
  const int tid = threadIdx.x;
  const int NT  = blockDim.x;
  const float* xg = x + (size_t)b * (3 * 64 * 128);

  for (int i = tid; i < 162; i += NT) w1L[i] = w1[i];
  for (int i = tid; i < 54;  i += NT) w2L[i] = w2[i];
  if (tid < 6)  b1L[tid] = b1[tid];
  if (tid == 0) b2L[0]   = b2[0];
  __syncthreads();

  // conv1 + leaky into midL: local rows lt=0..13 (global conv1 t = t0+lt)
  for (int i = tid; i < 6 * 14 * 126; i += NT) {
    int oc = i / (14 * 126);
    int r  = i % (14 * 126);
    int lt = r / 126;
    int s  = r % 126;
    int t  = t0 + lt;
    float acc = b1L[oc];
#pragma unroll
    for (int ic = 0; ic < 3; ++ic)
#pragma unroll
      for (int dt = 0; dt < 3; ++dt) {
        const float* xr = xg + ic * (64 * 128) + (t + dt) * 128 + s;
        const float* wr = w1L + ((oc * 3 + ic) * 3 + dt) * 3;
        acc += xr[0] * wr[0] + xr[1] * wr[1] + xr[2] * wr[2];
      }
    midL[i] = acc >= 0.f ? acc : 0.01f * acc;
  }
  __syncthreads();

  // conv2 + leaky -> f16 seq rows t0..t0+11, zero-padded to 128 cols
  _Float16* sq = seq + (size_t)b * (T_SEQ * S_PAD);
  for (int i = tid; i < 12 * S_PAD; i += NT) {
    int lt = i >> 7;
    int s  = i & 127;
    float outv = 0.f;
    if (s < 124) {
      float acc = b2L[0];
#pragma unroll
      for (int oc = 0; oc < 6; ++oc)
#pragma unroll
        for (int dt = 0; dt < 3; ++dt) {
          const float* mr = midL + oc * (14 * 126) + (lt + dt) * 126 + s;
          const float* wr = w2L + (oc * 3 + dt) * 3;
          acc += mr[0] * wr[0] + mr[1] * wr[1] + mr[2] * wr[2];
        }
      outv = acc >= 0.f ? acc : 0.01f * acc;
    }
    sq[(t0 + lt) * S_PAD + s] = (_Float16)outv;
  }
}

// ---------------------------------------------------------------------------
// WMMA fragment helpers (CDNA5 wave32 layouts, ISA 7.12.2):
//   A 16x32 f16: lane -> row m=lane&15, half=lane>>4; per-lane K offsets
//                half*8 + {0..7} and half*8 + {16..23}  (two b128 LDS loads)
//   B 32x16 f16: lane -> col n=lane&15, same per-lane K mapping.
//   C/D 16x16 f32: VGPR v -> row v+8*half, col lane&15.
// ---------------------------------------------------------------------------
__device__ inline v16h ldsA(const _Float16* base, int rs, int m, int k0, int half) {
  const _Float16* p = base + m * rs + k0 + half * 8;
  v16h a;
#pragma unroll
  for (int i = 0; i < 8; ++i) a[i] = p[i];
#pragma unroll
  for (int i = 0; i < 8; ++i) a[i + 8] = p[16 + i];
  return a;
}

// B fragment of W^T where W is [Nrows][K] row-major (W^T[k][n] = W[n*K+k]).
__device__ inline v16h bfragT(const float* __restrict__ W, int n, int K, int Nmax,
                              int k0, int half) {
  v16h b;
#pragma unroll
  for (int e = 0; e < 16; ++e) {
    int k = k0 + half * 8 + e + ((e >= 8) ? 8 : 0);
    float v = (n < Nmax && k < K) ? W[n * K + k] : 0.f;
    b[e] = (_Float16)v;
  }
  return b;
}

// ---------------------------------------------------------------------------
// Kernel 2: GRU recurrence + FC heads + squashed-Gaussian epilogue.
// One block per 16-sample batch tile; 25 waves = 25 gate-column tiles.
// ---------------------------------------------------------------------------
__global__ __launch_bounds__(800) void actor_kernel(
    const _Float16* __restrict__ seq,  // [B,60,128] f16
    const float* __restrict__ w_in,    // [B,129]
    const float* __restrict__ eps,     // [B,129]
    const float* __restrict__ Wih,     // [387,124]
    const float* __restrict__ Whh,     // [387,129]
    const float* __restrict__ bih,     // [387]
    const float* __restrict__ bhh,     // [387]
    const float* __restrict__ Wmu,     // [129,258]
    const float* __restrict__ bmu,     // [129]
    const float* __restrict__ Wls,     // [129,258]
    const float* __restrict__ bls,     // [129]
    float* __restrict__ out)           // [4096*129 pi | 4096 log_pi]
{
  extern __shared__ char smem[];
  _Float16* xA    = (_Float16*)smem;             // [16][128] x_t tile
  _Float16* hA    = xA + 16 * S_PAD;             // [16][160] h (f16, A layout)
  _Float16* featA = hA + 16 * HPAD;              // [16][288] feat (f16)
  float* hF    = (float*)(featA + 16 * FPAD);    // [16][132] h (f32)
  float* giL   = hF + 16 * 132;                  // [16][400] x@Wih^T
  float* ghL   = giL + 16 * GPAD;                // [16][400] h@Whh^T
  float* muL   = ghL + 16 * GPAD;                // [16][144]
  float* lsL   = muL + 16 * 144;                 // [16][144]
  float* logpiL = lsL + 16 * 144;                // [16]
  float* psumL  = logpiL + 16;                   // [16]
  float* bihL   = psumL + 16;                    // [388]
  float* bhhL   = bihL + 388;                    // [388]

  const int tid  = threadIdx.x;
  const int wid  = tid >> 5;        // wave id 0..24
  const int lane = tid & 31;
  const int m    = lane & 15;
  const int half = lane >> 4;
  const size_t b0 = (size_t)blockIdx.x * 16;

  // init: h = 0 (incl. f16 K-pad cols 129..159), stage gate biases
  for (int i = tid; i < 16 * HPAD; i += blockDim.x) hA[i] = (_Float16)0.f;
  for (int i = tid; i < 16 * 132;  i += blockDim.x) hF[i] = 0.f;
  for (int i = tid; i < G3; i += blockDim.x) { bihL[i] = bih[i]; bhhL[i] = bhh[i]; }

  // resident weight B-fragments for this wave's 16 gate columns
  const int ncol = 16 * wid + m;
  v16h Bih[4], Bhh[5];
  if (wid < NTILE) {
#pragma unroll
    for (int kt = 0; kt < 4; ++kt) Bih[kt] = bfragT(Wih, ncol, 124, G3, 32 * kt, half);
#pragma unroll
    for (int kt = 0; kt < 5; ++kt) Bhh[kt] = bfragT(Whh, ncol, HID, G3, 32 * kt, half);
  }
  __syncthreads();

  for (int t = 0; t < T_SEQ; ++t) {
    // stage x_t tile [16][128] f16 into LDS
    for (int i = tid; i < 16 * S_PAD; i += blockDim.x) {
      int r = i >> 7, s = i & 127;
      xA[i] = seq[(b0 + r) * (size_t)(T_SEQ * S_PAD) + t * S_PAD + s];
    }
    if (t + 1 < T_SEQ && tid < 16)
      __builtin_prefetch(&seq[(b0 + tid) * (size_t)(T_SEQ * S_PAD) + (t + 1) * S_PAD], 0, 0);
    __syncthreads();

    if (wid < NTILE) {
      v8f ai = {};
      v8f ah = {};
#pragma unroll
      for (int kt = 0; kt < 4; ++kt) {
        v16h a = ldsA(xA, S_PAD, m, 32 * kt, half);
        ai = __builtin_amdgcn_wmma_f32_16x16x32_f16(false, a, false, Bih[kt],
                                                    (short)0, ai, false, false);
      }
#pragma unroll
      for (int kt = 0; kt < 5; ++kt) {
        v16h a = ldsA(hA, HPAD, m, 32 * kt, half);
        ah = __builtin_amdgcn_wmma_f32_16x16x32_f16(false, a, false, Bhh[kt],
                                                    (short)0, ah, false, false);
      }
#pragma unroll
      for (int v = 0; v < 8; ++v) {
        int row = v + 8 * half;
        giL[row * GPAD + ncol] = ai[v];
        ghL[row * GPAD + ncol] = ah[v];
      }
    }
    __syncthreads();

    // GRU cell elementwise update
    for (int i = tid; i < 16 * HID; i += blockDim.x) {
      int r = i / HID, hh = i % HID;
      float ir  = giL[r * GPAD + hh]           + bihL[hh];
      float iz  = giL[r * GPAD + HID + hh]     + bihL[HID + hh];
      float inn = giL[r * GPAD + 2 * HID + hh] + bihL[2 * HID + hh];
      float hr  = ghL[r * GPAD + hh]           + bhhL[hh];
      float hz  = ghL[r * GPAD + HID + hh]     + bhhL[HID + hh];
      float hn  = ghL[r * GPAD + 2 * HID + hh] + bhhL[2 * HID + hh];
      float rg = 1.f / (1.f + __expf(-(ir + hr)));
      float zg = 1.f / (1.f + __expf(-(iz + hz)));
      float ng = tanhf(inn + rg * hn);
      float hnew = (1.f - zg) * ng + zg * hF[r * 132 + hh];
      hF[r * 132 + hh] = hnew;
      hA[r * HPAD + hh] = (_Float16)hnew;
    }
    __syncthreads();
  }

  // feat = [leaky(h), w, 0-pad] as f16 A operand
  for (int i = tid; i < 16 * FPAD; i += blockDim.x) {
    int r = i / FPAD, k = i % FPAD;
    float v;
    if (k < HID)        { float g = hF[r * 132 + k]; v = g >= 0.f ? g : 0.01f * g; }
    else if (k < 2*HID) { v = w_in[(b0 + r) * HID + (k - HID)]; }
    else                { v = 0.f; }
    featA[i] = (_Float16)v;
  }
  __syncthreads();

  // FC heads: waves 0..8 -> mu tiles, waves 9..17 -> log_std tiles
  if (wid < 18) {
    const float* W  = (wid < 9) ? Wmu : Wls;
    const float* bb = (wid < 9) ? bmu : bls;
    float* dst      = (wid < 9) ? muL : lsL;
    int jt = (wid < 9) ? wid : wid - 9;
    int oc = 16 * jt + m;
    v16h Bf[9];
#pragma unroll
    for (int kt = 0; kt < 9; ++kt) Bf[kt] = bfragT(W, oc, 2 * HID, HID, 32 * kt, half);
    v8f acc = {};
#pragma unroll
    for (int kt = 0; kt < 9; ++kt) {
      v16h a = ldsA(featA, FPAD, m, 32 * kt, half);
      acc = __builtin_amdgcn_wmma_f32_16x16x32_f16(false, a, false, Bf[kt],
                                                   (short)0, acc, false, false);
    }
    float bias = (oc < HID) ? bb[oc] : 0.f;
#pragma unroll
    for (int v = 0; v < 8; ++v) dst[(v + 8 * half) * 144 + oc] = acc[v] + bias;
  }
  if (tid < 16) { logpiL[tid] = 0.f; psumL[tid] = 0.f; }
  __syncthreads();

  // squashed-Gaussian epilogue with LDS f32 atomic row reductions
  const float HALF_LOG_2PI = 0.9189385332046727f;
  for (int i = tid; i < 16 * HID; i += blockDim.x) {
    int r = i / HID, hh = i % HID;
    float mu = muL[r * 144 + hh];
    float ls = fminf(fmaxf(lsL[r * 144 + hh], -20.f), 2.f);
    float sd = __expf(ls);
    float e  = eps[(b0 + r) * HID + hh];
    float xs = mu + sd * e;
    float pi = tanhf(xs);
    float d  = (xs - mu) / sd;
    float lp = -0.5f * d * d - ls - HALF_LOG_2PI - __logf(1.f - pi * pi + 1e-6f);
    atomicAdd(&logpiL[r], lp);           // ds_add_f32
    atomicAdd(&psumL[r], pi + 1.f);      // ds_add_f32
    muL[r * 144 + hh] = pi;              // stash tanh for normalize pass
  }
  __syncthreads();

  for (int i = tid; i < 16 * HID; i += blockDim.x) {
    int r = i / HID, hh = i % HID;
    out[(b0 + r) * HID + hh] = (muL[r * 144 + hh] + 1.f) / psumL[r];
  }
  if (tid < 16) out[(size_t)B_TOT * HID + b0 + tid] = logpiL[tid];
}

// ---------------------------------------------------------------------------
extern "C" void kernel_launch(void* const* d_in, const int* in_sizes, int n_in,
                              void* d_out, int out_size, void* d_ws, size_t ws_size,
                              hipStream_t stream) {
  const float* x       = (const float*)d_in[0];
  const float* w       = (const float*)d_in[1];
  const float* eps     = (const float*)d_in[2];
  const float* conv1_w = (const float*)d_in[3];
  const float* conv1_b = (const float*)d_in[4];
  const float* conv2_w = (const float*)d_in[5];
  const float* conv2_b = (const float*)d_in[6];
  const float* gru_Wih = (const float*)d_in[7];
  const float* gru_Whh = (const float*)d_in[8];
  const float* gru_bih = (const float*)d_in[9];
  const float* gru_bhh = (const float*)d_in[10];
  const float* fc_mu_w = (const float*)d_in[11];
  const float* fc_mu_b = (const float*)d_in[12];
  const float* fc_ls_w = (const float*)d_in[13];
  const float* fc_ls_b = (const float*)d_in[14];
  float* out = (float*)d_out;

  _Float16* seq = (_Float16*)d_ws;     // [4096][60][128] f16, ~60 MB

  // Kernel 1: conv pipeline, 5 t-chunks of 12 rows each.
  size_t lds1 = (size_t)(6 * 14 * 126 + 162 + 6 + 54 + 1) * sizeof(float);
  conv_kernel<<<dim3(B_TOT, 5), 256, lds1, stream>>>(
      x, conv1_w, conv1_b, conv2_w, conv2_b, seq);

  // Kernel 2: GRU + FC heads + epilogue, 16-sample tiles.
  size_t lds2 = (size_t)(16 * S_PAD + 16 * HPAD + 16 * FPAD) * sizeof(_Float16) +
                (size_t)(16 * 132 + 2 * 16 * GPAD + 2 * 16 * 144 + 32 + 2 * 388) *
                    sizeof(float);
  hipFuncSetAttribute((const void*)actor_kernel,
                      hipFuncAttributeMaxDynamicSharedMemorySize, (int)lds2);
  actor_kernel<<<B_TOT / 16, 800, lds2, stream>>>(
      seq, w, eps, gru_Wih, gru_Whh, gru_bih, gru_bhh,
      fc_mu_w, fc_mu_b, fc_ls_w, fc_ls_b, out);
}
